// RNN_69423851373165
// MI455X (gfx1250) — compile-verified, matrix operand order checked by reading, not run on hardware
//
#include <hip/hip_runtime.h>
#include <hip/hip_bf16.h>
#include <math.h>
#include <stdint.h>

typedef __attribute__((ext_vector_type(16))) __bf16 v16bf;
typedef __attribute__((ext_vector_type(8)))  float  v8f;
typedef uint32_t u32x4 __attribute__((ext_vector_type(4)));
typedef int      i32x4 __attribute__((ext_vector_type(4)));
typedef int      i32x8 __attribute__((ext_vector_type(8)));

#if defined(__has_builtin)
#if __has_builtin(__builtin_amdgcn_tensor_load_to_lds) && \
    __has_builtin(__builtin_amdgcn_s_wait_tensorcnt)
#define USE_TDM 1
#endif
#endif

union FragU {
    uint32_t u[8];
    uint4    q[2];
    v16bf    v;
};

__device__ __forceinline__ uint16_t f2bf(float f) {
    uint32_t x = __float_as_uint(f);
    x += 0x7fffu + ((x >> 16) & 1u);   // round to nearest even
    return (uint16_t)(x >> 16);
}

__device__ __forceinline__ uint32_t pack2bf(float lo, float hi) {
    return (uint32_t)f2bf(lo) | ((uint32_t)f2bf(hi) << 16);
}

// ---------------------------------------------------------------------------
// Elementwise f32 -> bf16 (RNE) with scale, 4 elements per thread-iter.
// ---------------------------------------------------------------------------
__global__ void cvt4_bf16_kernel(const float4* __restrict__ src,
                                 uint2* __restrict__ dst,
                                 float scale, int n4) {
    int i = blockIdx.x * blockDim.x + threadIdx.x;
    int stride = gridDim.x * blockDim.x;
    for (; i < n4; i += stride) {
        float4 s = src[i];
        uint2 d;
        d.x = pack2bf(s.x * scale, s.y * scale);
        d.y = pack2bf(s.z * scale, s.w * scale);
        dst[i] = d;
    }
}

// ---------------------------------------------------------------------------
// GEMM: C[M x N] (f32) = alpha * A[M x K] (bf16 rm) @ W[N x K]^T (bf16)
// 128x128 block tile, BK=32, 8 waves; TDM double-buffered tile staging.
// ---------------------------------------------------------------------------
#define BM 128
#define BN 128
#define BK 32
#define LDS_STR 40   // uint16 units; 80B row stride (16B aligned, conflict-free)

__device__ __forceinline__ uint32_t lds_addr_of(const void* p) {
    return (uint32_t)(uintptr_t)p;   // generic shared addr: low 32 bits = LDS offset
}

// Issue one TDM 2D tile load: BK x BM (elems) bf16 tile from a row-major
// tensor with row stride `stride_d0` elements, padded in LDS to LDS_STR rows.
__device__ __forceinline__ void tdm_load_tile(uint32_t lds_byte_addr,
                                              const void* gtile,
                                              uint32_t tensor_d0,
                                              uint32_t tensor_d1,
                                              uint32_t stride_d0) {
#ifdef USE_TDM
    const uint64_t ga = (uint64_t)(uintptr_t)gtile;
    u32x4 g0;
    g0[0] = 1u;                                             // count=1, user D#
    g0[1] = lds_byte_addr;                                  // lds_addr
    g0[2] = (uint32_t)ga;                                   // global_addr[31:0]
    g0[3] = (uint32_t)((ga >> 32) & 0x01FFFFFFu)            // global_addr[56:32]
            | 0x80000000u;                                  // type=2 (image)
    i32x8 g1;
    g1[0] = (int)(0x00010000u        // data_size = 1 -> 2 bytes/elem
                | 0x00100000u        // pad_enable
                | (3u << 22)         // pad_interval: every 16 dwords (64B row)
                | (3u << 25));       // pad_amount: 4 dwords (16B)
    g1[1] = (int)((tensor_d0 & 0xFFFFu) << 16);                          // dim0 lo16
    g1[2] = (int)(((tensor_d0 >> 16) & 0xFFFFu) |
                  ((tensor_d1 & 0xFFFFu) << 16));                        // dim0 hi / dim1 lo
    g1[3] = (int)(((tensor_d1 >> 16) & 0xFFFFu) |
                  ((uint32_t)BK << 16));                                 // dim1 hi / tile_dim0
    g1[4] = (int)BM;                 // tile_dim1 = 128 rows, tile_dim2 = 0
    g1[5] = (int)stride_d0;          // tensor_dim0_stride[31:0]
    g1[6] = 0;
    g1[7] = 0;
    const i32x4 z4 = {0, 0, 0, 0};
    const i32x8 z8 = {0, 0, 0, 0, 0, 0, 0, 0};
    // 6-arg toolchain form: (g0, g1, g2, g3, g4, cpol)
    __builtin_amdgcn_tensor_load_to_lds(g0, g1, z4, z4, z8, 0);
#else
    (void)lds_byte_addr; (void)gtile; (void)tensor_d0; (void)tensor_d1; (void)stride_d0;
#endif
}

__device__ __forceinline__ void tile_mma(const uint16_t* __restrict__ tA,
                                         const uint16_t* __restrict__ tB,
                                         int wrow, int wcol, int lr, int kb,
                                         v8f acc[4][2]) {
    FragU af[4], bf2[2];
#pragma unroll
    for (int mi = 0; mi < 4; mi++) {
        const int r = wrow * 64 + mi * 16 + lr;
        af[mi].q[0] = *(const uint4*)&tA[r * LDS_STR + kb];
        af[mi].q[1] = *(const uint4*)&tA[r * LDS_STR + 16 + kb];
    }
#pragma unroll
    for (int ni = 0; ni < 2; ni++) {
        const int r = wcol * 32 + ni * 16 + lr;
        bf2[ni].q[0] = *(const uint4*)&tB[r * LDS_STR + kb];
        bf2[ni].q[1] = *(const uint4*)&tB[r * LDS_STR + 16 + kb];
    }
#pragma unroll
    for (int mi = 0; mi < 4; mi++)
#pragma unroll
        for (int ni = 0; ni < 2; ni++)
            acc[mi][ni] = __builtin_amdgcn_wmma_f32_16x16x32_bf16(
                false, af[mi].v, false, bf2[ni].v,
                (short)0, acc[mi][ni], false, false);
}

__global__ __launch_bounds__(256)
void gemm_bf16_kernel(const uint16_t* __restrict__ A,
                      const uint16_t* __restrict__ W,
                      float* __restrict__ C,
                      int M, int N, int K, float alpha) {
#ifdef USE_TDM
    __shared__ uint16_t sA[2][BM * LDS_STR];
    __shared__ uint16_t sB[2][BN * LDS_STR];
#else
    __shared__ uint16_t sA[1][BM * LDS_STR];
    __shared__ uint16_t sB[1][BN * LDS_STR];
#endif

    const int tid  = threadIdx.x;
    const int lane = tid & 31;
    const int wv   = tid >> 5;       // 0..7
    const int wrow = wv & 1;         // 2 row groups of 64
    const int wcol = wv >> 1;        // 4 col groups of 32

    const int row0 = blockIdx.y * BM;
    const int col0 = blockIdx.x * BN;

    const int lr = lane & 15;
    const int kb = (lane >> 4) << 3; // 0 or 8 (K sub-offset per half-wave)

    v8f acc[4][2];
#pragma unroll
    for (int mi = 0; mi < 4; mi++)
#pragma unroll
        for (int ni = 0; ni < 2; ni++)
            acc[mi][ni] = (v8f){0.f, 0.f, 0.f, 0.f, 0.f, 0.f, 0.f, 0.f};

#ifdef USE_TDM
    const int nk = K / BK;
    if (wv == 0) {   // wave-uniform branch: only wave 0 issues TDM descriptors
        tdm_load_tile(lds_addr_of(sA[0]), &A[(size_t)row0 * K], K, M, K);
        tdm_load_tile(lds_addr_of(sB[0]), &W[(size_t)col0 * K], K, N, K);
    }
    for (int ks = 0; ks < nk; ks++) {
        if (wv == 0) {
            if (ks + 1 < nk) {   // issue next stage, then drain previous one
                tdm_load_tile(lds_addr_of(sA[(ks + 1) & 1]),
                              &A[(size_t)row0 * K + (size_t)(ks + 1) * BK], K, M, K);
                tdm_load_tile(lds_addr_of(sB[(ks + 1) & 1]),
                              &W[(size_t)col0 * K + (size_t)(ks + 1) * BK], K, N, K);
                __builtin_amdgcn_s_wait_tensorcnt(2);
            } else {
                __builtin_amdgcn_s_wait_tensorcnt(0);
            }
        }
        __syncthreads();                       // stage ks visible to all waves
        tile_mma(sA[ks & 1], sB[ks & 1], wrow, wcol, lr, kb, acc);
        __syncthreads();                       // done reading before overwrite
    }
#else
    for (int k0 = 0; k0 < K; k0 += BK) {
#pragma unroll
        for (int c = tid; c < (BM * BK / 8); c += 256) {
            const int r = c >> 2;
            const int s = (c & 3) << 3;
            *(uint4*)&sA[0][r * LDS_STR + s] =
                *(const uint4*)&A[(size_t)(row0 + r) * K + k0 + s];
            *(uint4*)&sB[0][r * LDS_STR + s] =
                *(const uint4*)&W[(size_t)(col0 + r) * K + k0 + s];
            const int kn = (k0 + BK < K) ? (k0 + BK) : k0;
            __builtin_prefetch(&A[(size_t)(row0 + r) * K + kn + s], 0, 0);
            __builtin_prefetch(&W[(size_t)(col0 + r) * K + kn + s], 0, 0);
        }
        __syncthreads();
        tile_mma(sA[0], sB[0], wrow, wcol, lr, kb, acc);
        __syncthreads();
    }
#endif

    // Epilogue: D layout -> row = base + v + 8*(lane>=16), col = base + lane%16
    const int rsub = (lane >> 4) << 3;
#pragma unroll
    for (int mi = 0; mi < 4; mi++)
#pragma unroll
        for (int ni = 0; ni < 2; ni++)
#pragma unroll
            for (int v = 0; v < 8; v++) {
                const int row = row0 + wrow * 64 + mi * 16 + v + rsub;
                const int col = col0 + wcol * 32 + ni * 16 + lr;
                C[(size_t)row * N + col] = acc[mi][ni][v] * alpha;
            }
}

// ---------------------------------------------------------------------------
// Recurrence: h_t = tanh(h_{t-1} @ W_n + x_t), one block per head.
// 4 waves; wave wv owns output columns [wv*16, wv*16+16). M=4 (batch) padded
// to 16; x_t loads straight into the WMMA f32 C accumulator.
// ---------------------------------------------------------------------------
#define HB_STR 72    // uint16 units; 144B stride (16B aligned, conflict-free)

__global__ __launch_bounds__(128)
void recurrence_kernel(const float* __restrict__ x,      // (B*S) x D f32
                       const uint16_t* __restrict__ wst, // N x 64 x 64 bf16 (scaled)
                       uint16_t* __restrict__ y,         // (B*S) x D bf16
                       int S, int D) {
    const int n    = blockIdx.x;
    const int tid  = threadIdx.x;
    const int lane = tid & 31;
    const int wv   = tid >> 5;       // 0..3 column tile
    const int lr   = lane & 15;
    const int kb   = (lane >> 4) << 3;

    __shared__ uint16_t hbuf[16 * HB_STR];
    for (int i = tid; i < 16 * HB_STR; i += 128) hbuf[i] = 0;

    // Hoist W_n B-fragments into registers (constant over t).
    // B[kdim=h][col=c] = wst[n][h][c]; lane owns col c = wv*16 + lr.
    FragU bw[2];
    const uint16_t* wn = wst + (size_t)n * 64 * 64;
    const int c = wv * 16 + lr;
#pragma unroll
    for (int kc = 0; kc < 2; kc++)
#pragma unroll
        for (int v = 0; v < 8; v++) {
            const int h0 = kc * 32 + ((v & 4) ? 16 : 0) + kb + ((v & 3) << 1);
            const uint32_t lo = wn[h0 * 64 + c];
            const uint32_t hi = wn[(h0 + 1) * 64 + c];
            bw[kc].u[v] = lo | (hi << 16);
        }
    __syncthreads();

    const size_t colg = (size_t)n * 64 + c;   // global column of this lane

    for (int t = 0; t < S; t++) {
        // C accumulator := x_t. Unconditional across the wave: lanes 0..15
        // feed rows b=0..3; lanes 16..31 land in rows 8..11, never stored.
        v8f acc = (v8f){0.f, 0.f, 0.f, 0.f, 0.f, 0.f, 0.f, 0.f};
#pragma unroll
        for (int b = 0; b < 4; b++)
            acc[b] = x[((size_t)b * S + t) * D + colg];
        const int tp = (t + 8 < S) ? (t + 8) : t;   // clamped, branch-free
        __builtin_prefetch(&x[(size_t)tp * D + colg], 0, 1);

        // A fragments (16 rows x 64 K) from bf16 h buffer
        FragU a0, a1;
        a0.q[0] = *(const uint4*)&hbuf[lr * HB_STR + kb];
        a0.q[1] = *(const uint4*)&hbuf[lr * HB_STR + 16 + kb];
        a1.q[0] = *(const uint4*)&hbuf[lr * HB_STR + 32 + kb];
        a1.q[1] = *(const uint4*)&hbuf[lr * HB_STR + 48 + kb];

        acc = __builtin_amdgcn_wmma_f32_16x16x32_bf16(
            false, a0.v, false, bw[0].v, (short)0, acc, false, false);
        acc = __builtin_amdgcn_wmma_f32_16x16x32_bf16(
            false, a1.v, false, bw[1].v, (short)0, acc, false, false);

        __syncthreads();   // everyone done reading hbuf
        if (lane < 16) {
#pragma unroll
            for (int b = 0; b < 4; b++) {
                const uint16_t hb = f2bf(tanhf(acc[b]));
                hbuf[b * HB_STR + c] = hb;
                y[((size_t)b * S + t) * D + colg] = hb;
            }
        }
        __syncthreads();   // hbuf ready for next step
    }
}

// ---------------------------------------------------------------------------
extern "C" void kernel_launch(void* const* d_in, const int* in_sizes, int n_in,
                              void* d_out, int out_size, void* d_ws, size_t ws_size,
                              hipStream_t stream) {
    (void)in_sizes; (void)n_in; (void)out_size; (void)ws_size;

    const float* in    = (const float*)d_in[0];  // (B,S,D)
    const float* w_in  = (const float*)d_in[1];  // (D,D)
    const float* wst   = (const float*)d_in[2];  // (N,H,H)
    const float* w_out = (const float*)d_in[3];  // (D,D)
    float* out = (float*)d_out;

    const int B = 4, S = 4096, D = 2048, N = 32, H = 64;
    const int M = B * S;                          // 16384
    const float factor = 1.0f / sqrtf((float)(D + H));

    char* ws = (char*)d_ws;
    uint16_t* in_bf   = (uint16_t*)(ws);                 // 67,108,864 B
    uint16_t* win_bf  = (uint16_t*)(ws + 67108864);      //  8,388,608 B
    uint16_t* wout_bf = (uint16_t*)(ws + 75497472);      //  8,388,608 B
    uint16_t* wst_bf  = (uint16_t*)(ws + 83886080);      //    262,144 B
    float*    x_f32   = (float*)   (ws + 84148224);      // 134,217,728 B
    uint16_t* y_bf    = (uint16_t*)(ws + 218365952);     // 67,108,864 B

    // 1) f32 -> bf16 conversions (w_state pre-scaled by factor)
    {
        int n4 = (M * D) / 4;
        cvt4_bf16_kernel<<<(n4 + 1023) / 1024, 256, 0, stream>>>(
            (const float4*)in, (uint2*)in_bf, 1.0f, n4);
        n4 = (D * D) / 4;
        cvt4_bf16_kernel<<<(n4 + 1023) / 1024, 256, 0, stream>>>(
            (const float4*)w_in, (uint2*)win_bf, 1.0f, n4);
        cvt4_bf16_kernel<<<(n4 + 1023) / 1024, 256, 0, stream>>>(
            (const float4*)w_out, (uint2*)wout_bf, 1.0f, n4);
        n4 = (N * H * H) / 4;
        cvt4_bf16_kernel<<<(n4 + 1023) / 1024, 256, 0, stream>>>(
            (const float4*)wst, (uint2*)wst_bf, factor, n4);
    }

    // 2) x = factor * (input @ w_in^T)
    {
        dim3 grid(D / BN, M / BM);
        gemm_bf16_kernel<<<grid, 256, 0, stream>>>(in_bf, win_bf, x_f32,
                                                   M, D, D, factor);
    }

    // 3) sequential recurrence per head
    recurrence_kernel<<<N, 128, 0, stream>>>(x_f32, wst_bf, y_bf, S, D);

    // 4) out = y @ w_out^T
    {
        dim3 grid(D / BN, M / BM);
        gemm_bf16_kernel<<<grid, 256, 0, stream>>>(y_bf, wout_bf, out,
                                                   M, D, D, 1.0f);
    }
}
